// MAMBA_BayesMAGAC_6811818131893
// MI455X (gfx1250) — matrix operand order repaired and checked by reading, not compiled, not run on hardware
//
#include <hip/hip_runtime.h>
#include <hip/hip_bf16.h>
#include <math.h>

typedef __attribute__((ext_vector_type(16))) _Float16 v16h;
typedef __attribute__((ext_vector_type(8)))  float    v8f;
typedef _Float16 h16;

#define B_  32
#define L_  512
#define D_  128
#define E_  64
#define HS_ 64
#define NT_ 16384   // B_*L_

// ---------------------------------------------------------------- utilities
__device__ __forceinline__ float warp_sum(float v) {
#pragma unroll
  for (int off = 16; off >= 1; off >>= 1) v += __shfl_xor(v, off, 32);
  return v;
}

__global__ void k_f2h(const float* __restrict__ s, h16* __restrict__ d, int n) {
  int i = blockIdx.x * blockDim.x + threadIdx.x;
  if (i < n) d[i] = (h16)s[i];
}

__global__ void k_copy(const float* __restrict__ s, float* __restrict__ d, int n) {
  int i = blockIdx.x * blockDim.x + threadIdx.x;
  if (i < n) d[i] = s[i];
}

__global__ void k_zero(float* __restrict__ d, int n) {
  int i = blockIdx.x * blockDim.x + threadIdx.x;
  if (i < n) d[i] = 0.0f;
}

// zero-pad columns while converting to f16 (for xproj 132 -> 144)
__global__ void k_pad16(const float* __restrict__ s, h16* __restrict__ d,
                        int rows, int sc, int dc) {
  int i = blockIdx.x * blockDim.x + threadIdx.x;
  if (i >= rows * dc) return;
  int r = i / dc, c = i - r * dc;
  d[i] = (h16)((c < sc) ? s[r * sc + c] : 0.0f);
}

// ---------------------------------------------------------------- WMMA GEMM
// D = alpha * (A f16[MxK] @ B f16[KxN]) + beta * C ; optional f16 mirror D16.
// Block: 256 threads = 8 waves; block tile 64(M) x 64(N); wave = two 16x16
// m-subtiles sharing one B fragment. A 64x32 and B 32x64 tiles are staged
// into LDS per k-step with GLOBAL_LOAD_ASYNC_TO_LDS_B128 (ASYNCcnt), then
// WMMA fragments are read from LDS. Requirements (all call sites comply):
// M % 64 == 0, N % 16 == 0, K % 32 == 0, 16B-aligned rows.
__global__ void k_gemm(const h16* __restrict__ A, int lda,
                       const h16* __restrict__ Bm, int ldb,
                       const float* __restrict__ C,
                       float* __restrict__ D, h16* __restrict__ D16, int ldc,
                       int M, int N, int K, float alpha, float beta)
{
  __shared__ h16 As[64 * 32];   // 4 KB
  __shared__ h16 Bs[32 * 64];   // 4 KB
  int tid = threadIdx.x, wave = tid >> 5, lane = tid & 31;
  int m_blk = blockIdx.x * 64, n_blk = blockIdx.y * 64;
  int n0 = n_blk + (wave & 3) * 16;
  int m0 = m_blk + (wave >> 2) * 16;
  // wave-uniform predicate forced into an SGPR -> scalar branch, no exec
  // masking around WMMA (EXEC must stay all-ones).
  int active = __builtin_amdgcn_readfirstlane((int)(n0 < N));

  // staging coordinates: thread copies one 16B chunk of A and one of B
  int sa_row = tid >> 2, sa_seg = tid & 3;   // A: 64 rows x 4 segs(8 halves)
  int sb_row = tid >> 3, sb_seg = tid & 7;   // B: 32 rows x 8 segs(8 halves)
  unsigned la = (unsigned)(size_t)&As[sa_row * 32 + sa_seg * 8];
  unsigned lb = (unsigned)(size_t)&Bs[sb_row * 64 + sb_seg * 8];
  bool bseg_ok = (n_blk + sb_seg * 8) < N;

  v8f acc0 = {}, acc1 = {};
  int koff = (lane & 16) ? 8 : 0;            // A: K-half select by lane group
  int r0 = (wave >> 2) * 16 + (lane & 15);   // A row within 64-row tile
  int noff = (wave & 3) * 16;                // B column offset within tile

  for (int k0 = 0; k0 < K; k0 += 32) {
    // ---- async stage global -> LDS (tracked by ASYNCcnt)
    {
      unsigned long long ga =
          (unsigned long long)(A + (size_t)(m_blk + sa_row) * lda + k0 + sa_seg * 8);
      asm volatile("global_load_async_to_lds_b128 %0, %1, off"
                   :: "v"(la), "v"(ga) : "memory");
    }
    if (bseg_ok) {
      unsigned long long gb =
          (unsigned long long)(Bm + (size_t)(k0 + sb_row) * ldb + n_blk + sb_seg * 8);
      asm volatile("global_load_async_to_lds_b128 %0, %1, off"
                   :: "v"(lb), "v"(gb) : "memory");
    }
    asm volatile("s_wait_asynccnt 0x0" ::: "memory");
    __syncthreads();

    if (active) {
      // B fragment: lane%16 = column noff+n, halves j -> K rows {0..7,16..23}/{8..15,24..31}
      const h16* bp = &Bs[koff * 64 + noff + (lane & 15)];
      v16h bf;
#pragma unroll
      for (int j = 0; j < 16; ++j) bf[j] = bp[((j & 7) + ((j >> 3) * 16)) * 64];
      // A fragment: lane group selects K halves {0..7,16..23}/{8..15,24..31}
      const h16* ap0 = &As[r0 * 32 + koff];
      v16h af0;
#pragma unroll
      for (int j = 0; j < 8; ++j) { af0[j] = ap0[j]; af0[8 + j] = ap0[16 + j]; }
      acc0 = __builtin_amdgcn_wmma_f32_16x16x32_f16(false, af0, false, bf,
                                                    (short)0, acc0, false, false);
      const h16* ap1 = ap0 + 32 * 32;        // second m-subtile (+32 rows)
      v16h af1;
#pragma unroll
      for (int j = 0; j < 8; ++j) { af1[j] = ap1[j]; af1[8 + j] = ap1[16 + j]; }
      acc1 = __builtin_amdgcn_wmma_f32_16x16x32_f16(false, af1, false, bf,
                                                    (short)0, acc1, false, false);
    }
    __syncthreads();
  }

  if (!active) return;
  int ncol = n0 + (lane & 15);
  int mrow = m0 + ((lane & 16) ? 8 : 0);
#pragma unroll
  for (int r = 0; r < 8; ++r) {
    size_t i0 = (size_t)(mrow + r) * ldc + ncol;
    float d0 = alpha * acc0[r];
    if (beta != 0.0f) d0 += beta * C[i0];
    D[i0] = d0;
    if (D16) D16[i0] = (h16)d0;
    size_t i1 = (size_t)(mrow + 32 + r) * ldc + ncol;
    float d1 = alpha * acc1[r];
    if (beta != 0.0f) d1 += beta * C[i1];
    D[i1] = d1;
    if (D16) D16[i1] = (h16)d1;
  }
}

// ------------------------------------------------------- mamba: conv + silu
// fwd: xc[p] = b + w0*x[p-2] + w1*x[p-1] + w2*x[p]
// bwd (flip/conv/flip): xc[p] = b + w0*x[p+2] + w1*x[p+1] + w2*x[p]
__global__ void k_conv_silu(const float* __restrict__ xz, const float* __restrict__ cw,
                            const float* __restrict__ cb, float* __restrict__ xp,
                            h16* __restrict__ xp16, int fwd)
{
  int idx = blockIdx.x * blockDim.x + threadIdx.x;
  if (idx >= NT_ * E_) return;
  int e = idx & 63, tok = idx >> 6;
  int b = tok >> 9, p = tok & 511;
  float acc = cb[e];
#pragma unroll
  for (int k = 0; k < 3; ++k) {
    int pp = fwd ? (p - 2 + k) : (p + 2 - k);
    if (pp >= 0 && pp < L_)
      acc += cw[e * 3 + k] * xz[((size_t)(b * L_ + pp)) * D_ + e];
  }
  float s = acc / (1.0f + expf(-acc));   // silu
  xp[idx] = s;
  xp16[idx] = (h16)s;
}

// ------------------------------------------------------------ selective scan
// One wave per (b,e); lane holds states hs = lane, lane+32. 512 serial steps.
__global__ void k_scan(const float* __restrict__ xdbl, const float* __restrict__ xp,
                       const float* __restrict__ xz, const float* __restrict__ dt_w,
                       const float* __restrict__ dt_b, const float* __restrict__ Alog,
                       const float* __restrict__ Dp, h16* __restrict__ y16, int fwd)
{
  int gtid = blockIdx.x * blockDim.x + threadIdx.x;
  int wid = gtid >> 5, lane = gtid & 31;
  if (wid >= B_ * E_) return;
  int b = wid >> 6, e = wid & 63;
  int hs0 = lane, hs1 = lane + 32;
  float A0 = -expf(Alog[e * HS_ + hs0]);
  float A1 = -expf(Alog[e * HS_ + hs1]);
  float dtb = dt_b[e];
  float w0 = dt_w[e], w1 = dt_w[64 + e], w2 = dt_w[128 + e], w3 = dt_w[192 + e];
  float Dpe = Dp[e];
  float s0 = 0.0f, s1 = 0.0f;
  for (int t = 0; t < L_; ++t) {
    int p = fwd ? t : (L_ - 1 - t);
    size_t tok = (size_t)b * L_ + p;
    // hide latency of the serial recurrence: prefetch next step's row
    int pn = fwd ? (p + 1) : (p - 1);
    if (pn >= 0 && pn < L_)
      __builtin_prefetch(xdbl + ((size_t)b * L_ + pn) * 144, 0, 3);
    const float* xr = xdbl + tok * 144;
    float dlt = dtb + w0 * xr[0] + w1 * xr[1] + w2 * xr[2] + w3 * xr[3];
    dlt = (dlt > 20.0f) ? dlt : log1pf(expf(dlt));          // softplus
    float xpv = xp[tok * E_ + e];
    float Bv0 = xr[4 + hs0], Bv1 = xr[4 + hs1];
    float Cv0 = xr[68 + hs0], Cv1 = xr[68 + hs1];
    s0 = expf(dlt * A0) * s0 + dlt * Bv0 * xpv;
    s1 = expf(dlt * A1) * s1 + dlt * Bv1 * xpv;
    float y = warp_sum(s0 * Cv0 + s1 * Cv1);
    if (lane == 0) {
      float res = xz[tok * D_ + 64 + e];
      float val = (y + xpv * Dpe) / (1.0f + expf(-res));    // * sigmoid(res)
      y16[tok * E_ + e] = (h16)val;
    }
  }
}

// ----------------------------------------------------------------- layernorm
// One wave per token (D=128 -> 4 values/lane). val = in (+ addA) (+ bias)
__global__ void k_ln(const float* __restrict__ in, const float* __restrict__ addA,
                     const float* __restrict__ bias, const float* __restrict__ g,
                     const float* __restrict__ bta, float* __restrict__ out,
                     h16* __restrict__ out16)
{
  int gtid = blockIdx.x * blockDim.x + threadIdx.x;
  int tok = gtid >> 5, lane = gtid & 31;
  if (tok >= NT_) return;
  size_t base = (size_t)tok * D_;
  float v[4];
#pragma unroll
  for (int j = 0; j < 4; ++j) {
    int d = lane + 32 * j;
    float t = in[base + d];
    if (addA) t += addA[base + d];
    if (bias) t += bias[d];
    v[j] = t;
  }
  float s = warp_sum(v[0] + v[1] + v[2] + v[3]);
  float mean = s * (1.0f / 128.0f);
  float q = 0.0f;
#pragma unroll
  for (int j = 0; j < 4; ++j) { float df = v[j] - mean; q += df * df; }
  q = warp_sum(q);
  float inv = rsqrtf(q * (1.0f / 128.0f) + 1e-5f);
#pragma unroll
  for (int j = 0; j < 4; ++j) {
    int d = lane + 32 * j;
    float o = (v[j] - mean) * inv * g[d] + bta[d];
    out[base + d] = o;
    if (out16) out16[base + d] = (h16)o;
  }
}

__global__ void k_bias_relu(const float* __restrict__ F1, const float* __restrict__ b1,
                            h16* __restrict__ G16, int n) {
  int i = blockIdx.x * blockDim.x + threadIdx.x;
  if (i >= n) return;
  float v = F1[i] + b1[i & 31];
  G16[i] = (h16)(v > 0.0f ? v : 0.0f);
}

// H[m, tok] = x[tok, m]  (f32 + f16)
__global__ void k_transpose(const float* __restrict__ x, float* __restrict__ H,
                            h16* __restrict__ H16) {
  int i = blockIdx.x * blockDim.x + threadIdx.x;
  if (i >= D_ * NT_) return;
  int m = i >> 14, tok = i & 16383;
  float v = x[(size_t)tok * D_ + m];
  H[i] = v;
  H16[i] = (h16)v;
}

// ------------------------------------------------------------------- head
__global__ void k_Ag(const float* __restrict__ psi_emb, const float* __restrict__ psi_s,
                     float* __restrict__ Ag) {
  int n = blockIdx.x, m = threadIdx.x;
  __shared__ float sh[128];
  float d2 = 0.0f;
#pragma unroll
  for (int d = 0; d < 10; ++d) {
    float df = psi_emb[n * 10 + d] - psi_emb[m * 10 + d];
    d2 += df * df;
  }
  float z = expf(-psi_s[0] * d2);
  sh[m] = z; __syncthreads();
  for (int s = 64; s > 0; s >>= 1) { if (m < s) sh[m] = fmaxf(sh[m], sh[m + s]); __syncthreads(); }
  float mx = sh[0]; __syncthreads();
  float ez = expf(z - mx);
  sh[m] = ez; __syncthreads();
  for (int s = 64; s > 0; s >>= 1) { if (m < s) sh[m] += sh[m + s]; __syncthreads(); }
  Ag[n * 128 + m] = ez / sh[0];
}

__global__ void k_attn(const float* __restrict__ psi_emb, const float* __restrict__ Wq,
                       const float* __restrict__ Wk, const float* __restrict__ alpha_s,
                       const float* __restrict__ Ag, float* __restrict__ Aeff,
                       h16* __restrict__ Aeff16) {
  int h = blockIdx.x >> 7, n = blockIdx.x & 127, m = threadIdx.x;
  __shared__ float sh[128];
  float q[10], kk[10];
#pragma unroll
  for (int d = 0; d < 10; ++d) {
    float aq = 0.0f, ak = 0.0f;
    for (int dd = 0; dd < 10; ++dd) {
      aq += psi_emb[n * 10 + dd] * Wq[(dd * 4 + h) * 10 + d];
      ak += psi_emb[m * 10 + dd] * Wk[(dd * 4 + h) * 10 + d];
    }
    q[d] = aq; kk[d] = ak;
  }
  float logit = 0.0f;
#pragma unroll
  for (int d = 0; d < 10; ++d) logit += q[d] * kk[d];
  logit *= rsqrtf(10.0f);
  sh[m] = logit; __syncthreads();
  for (int s = 64; s > 0; s >>= 1) { if (m < s) sh[m] = fmaxf(sh[m], sh[m + s]); __syncthreads(); }
  float mx = sh[0]; __syncthreads();
  float e = expf(logit - mx);
  sh[m] = e; __syncthreads();
  for (int s = 64; s > 0; s >>= 1) { if (m < s) sh[m] += sh[m + s]; __syncthreads(); }
  float at = e / sh[0];
  float al = 1.0f / (1.0f + expf(-alpha_s[0]));
  float v = al * Ag[n * 128 + m] + (1.0f - al) * at;
  size_t o = (size_t)(h * 128 + n) * 128 + m;
  Aeff[o] = v;
  Aeff16[o] = (h16)v;
}

// Wf[h,n,k,l] = sum_d psi_emb[n,d] * F_w[h,d,k,l]
__global__ void k_Wf(const float* __restrict__ psi_emb, const float* __restrict__ Fw,
                     float* __restrict__ Wf) {
  int i = blockIdx.x * blockDim.x + threadIdx.x;
  if (i >= 4 * 128 * 4 * 512) return;
  int l = i & 511, k = (i >> 9) & 3, n = (i >> 11) & 127, h = i >> 18;
  float acc = 0.0f;
#pragma unroll
  for (int d = 0; d < 10; ++d)
    acc += psi_emb[n * 10 + d] * Fw[((size_t)(h * 10 + d) * 4 + k) * 512 + l];
  Wf[i] = acc;
}

// out[b,n] += mix[h] * ( sum_{k,l} Wf[h,n,k,l]*T_k[n, b*512+l] + bf[h,n] )
__global__ void k_head_reduce(const float* __restrict__ T0, const float* __restrict__ T1,
                              const float* __restrict__ T2, const float* __restrict__ T3,
                              const float* __restrict__ Wf, const float* __restrict__ psi_emb,
                              const float* __restrict__ f_b, const float* __restrict__ head_mix,
                              float* __restrict__ out, int h) {
  int gtid = blockIdx.x * blockDim.x + threadIdx.x;
  int wid = gtid >> 5, lane = gtid & 31;
  if (wid >= 128 * 32) return;
  int n = wid >> 5, b = wid & 31;
  const float* Ts[4] = { T0, T1, T2, T3 };
  size_t tbase = (size_t)n * NT_ + (size_t)b * L_;
  const float* wfb = Wf + ((size_t)(h * 128 + n) * 4) * 512;
  float acc = 0.0f;
  for (int k = 0; k < 4; ++k) {
    const float* T = Ts[k];
    const float* wf = wfb + k * 512;
    for (int l = lane; l < 512; l += 32) acc += wf[l] * T[tbase + l];
  }
  acc = warp_sum(acc);
  if (lane == 0) {
    float bf = 0.0f;
#pragma unroll
    for (int d = 0; d < 10; ++d) bf += psi_emb[n * 10 + d] * f_b[h * 10 + d];
    float m0 = head_mix[0], m1 = head_mix[1], m2 = head_mix[2], m3 = head_mix[3];
    float mx = fmaxf(fmaxf(m0, m1), fmaxf(m2, m3));
    float e0 = expf(m0 - mx), e1 = expf(m1 - mx), e2 = expf(m2 - mx), e3 = expf(m3 - mx);
    float mix = ((h == 0) ? e0 : (h == 1) ? e1 : (h == 2) ? e2 : e3) / (e0 + e1 + e2 + e3);
    out[b * 128 + n] += mix * (acc + bf);
  }
}

__global__ void k_pred(const float* __restrict__ out, const float* __restrict__ hw,
                       const float* __restrict__ hb, float* __restrict__ pred) {
  int gtid = blockIdx.x * blockDim.x + threadIdx.x;
  int wid = gtid >> 5, lane = gtid & 31;
  if (wid >= 32) return;
  float acc = 0.0f;
#pragma unroll
  for (int j = 0; j < 4; ++j) {
    int n = lane + 32 * j;
    acc += out[wid * 128 + n] * hw[n];
  }
  acc = warp_sum(acc);
  if (lane == 0) pred[wid] = acc + hb[0];
}

// ---------------------------------------------------------------- launcher
extern "C" void kernel_launch(void* const* d_in, const int* in_sizes, int n_in,
                              void* d_out, int out_size, void* d_ws, size_t ws_size,
                              hipStream_t stream) {
  (void)in_sizes; (void)n_in; (void)out_size; (void)ws_size;
  const float* x_in    = (const float*)d_in[0];
  const float* in_w    = (const float*)d_in[1];
  const float* conv_w  = (const float*)d_in[2];
  const float* conv_b  = (const float*)d_in[3];
  const float* xproj_w = (const float*)d_in[4];
  const float* dt_w    = (const float*)d_in[5];
  const float* dt_b    = (const float*)d_in[6];
  const float* Alog    = (const float*)d_in[7];
  const float* Dp      = (const float*)d_in[8];
  const float* out_w   = (const float*)d_in[9];
  const float* ln1_g   = (const float*)d_in[10];
  const float* ln1_b   = (const float*)d_in[11];
  const float* ln2_g   = (const float*)d_in[12];
  const float* ln2_b   = (const float*)d_in[13];
  const float* ffn_w1  = (const float*)d_in[14];
  const float* ffn_b1  = (const float*)d_in[15];
  const float* ffn_w2  = (const float*)d_in[16];
  const float* ffn_b2  = (const float*)d_in[17];
  const float* psi_emb = (const float*)d_in[18];
  const float* psi_s   = (const float*)d_in[19];
  const float* W_q     = (const float*)d_in[20];
  const float* W_k     = (const float*)d_in[21];
  const float* alpha_s = (const float*)d_in[22];
  const float* F_w     = (const float*)d_in[23];
  const float* f_b     = (const float*)d_in[24];
  const float* hmix    = (const float*)d_in[25];
  const float* head_w  = (const float*)d_in[26];
  const float* head_b  = (const float*)d_in[27];
  float* outp = (float*)d_out;

  char* base = (char*)d_ws;
  size_t off = 0;
  auto alloc = [&](size_t bytes) -> void* {
    off = (off + 255) & ~(size_t)255;
    void* p = base + off;
    off += bytes;
    return p;
  };
  const size_t NT = NT_;
  float* x_cur  = (float*)alloc(NT * 128 * 4);
  h16*   x16    = (h16*)  alloc(NT * 128 * 2);
  float* xz     = (float*)alloc(NT * 128 * 4);
  float* xp     = (float*)alloc(NT * 64 * 4);
  h16*   xp16   = (h16*)  alloc(NT * 64 * 2);
  float* xdbl   = (float*)alloc(NT * 144 * 4);
  h16*   y16    = (h16*)  alloc(NT * 64 * 2);
  float* Yacc   = (float*)alloc(NT * 128 * 4);
  float* Y3     = (float*)alloc(NT * 128 * 4);
  h16*   Y316   = (h16*)  alloc(NT * 128 * 2);
  float* F1     = (float*)alloc(NT * 32 * 4);
  h16*   G16    = (h16*)  alloc(NT * 32 * 2);
  float* F2     = (float*)alloc(NT * 128 * 4);
  h16*   in_w16 = (h16*)  alloc(6 * 128 * 128 * 2);
  h16*   xpr16  = (h16*)  alloc(6 * 64 * 144 * 2 + 256);
  h16*   ow16   = (h16*)  alloc(6 * 64 * 128 * 2);
  h16*   w1_16  = (h16*)  alloc(3 * 128 * 32 * 2);
  h16*   w2_16  = (h16*)  alloc(3 * 32 * 128 * 2);
  float* Hf     = (float*)alloc(128 * NT * 4);
  h16*   H16    = (h16*)  alloc(128 * NT * 2);
  float* T1     = (float*)alloc(128 * NT * 4);
  float* T2     = (float*)alloc(128 * NT * 4);
  float* T3     = (float*)alloc(128 * NT * 4);
  h16*   T1h    = (h16*)  alloc(128 * NT * 2);
  h16*   T2h    = (h16*)  alloc(128 * NT * 2);
  float* Ag     = (float*)alloc(128 * 128 * 4);
  float* Aeff   = (float*)alloc(4 * 128 * 128 * 4);
  h16*   Aeff16 = (h16*)  alloc(4 * 128 * 128 * 2);
  float* Wf     = (float*)alloc(4 * 128 * 4 * 512 * 4);
  float* outb   = (float*)alloc(32 * 128 * 4);

  dim3 b256(256);
  auto cdiv = [](int a, int b) { return (a + b - 1) / b; };

  // weight / input conversions (stateless, every call)
  k_f2h  <<<cdiv(6*128*128, 256), b256, 0, stream>>>(in_w,  in_w16, 6*128*128);
  k_pad16<<<cdiv(6*64*144, 256),  b256, 0, stream>>>(xproj_w, xpr16, 6*64, 132, 144);
  k_f2h  <<<cdiv(6*64*128, 256),  b256, 0, stream>>>(out_w, ow16,  6*64*128);
  k_f2h  <<<cdiv(3*128*32, 256),  b256, 0, stream>>>(ffn_w1, w1_16, 3*128*32);
  k_f2h  <<<cdiv(3*32*128, 256),  b256, 0, stream>>>(ffn_w2, w2_16, 3*32*128);
  k_copy <<<cdiv((int)(NT*128), 256), b256, 0, stream>>>(x_in, x_cur, NT*128);
  k_f2h  <<<cdiv((int)(NT*128), 256), b256, 0, stream>>>(x_in, x16, NT*128);

  for (int i = 0; i < 3; ++i) {
    k_copy<<<cdiv((int)(NT*128), 256), b256, 0, stream>>>(x_cur, Yacc, NT*128);
    for (int d = 0; d < 2; ++d) {
      int dir = 2 * i + d;
      int fwd = (d == 0);
      // xz = x @ in_w[dir]                       (16384 x 128 x 128)
      k_gemm<<<dim3(256, 2), b256, 0, stream>>>(x16, 128, in_w16 + dir*128*128, 128,
              nullptr, xz, nullptr, 128, NT_, 128, 128, 1.0f, 0.0f);
      k_conv_silu<<<cdiv(NT_*64, 256), b256, 0, stream>>>(xz, conv_w + dir*64*3,
              conv_b + dir*64, xp, xp16, fwd);
      // xdbl = xp @ xproj[dir] (padded to 144)   (16384 x 64 x 144)
      k_gemm<<<dim3(256, 3), b256, 0, stream>>>(xp16, 64, xpr16 + dir*64*144, 144,
              nullptr, xdbl, nullptr, 144, NT_, 144, 64, 1.0f, 0.0f);
      k_scan<<<cdiv(32*64*32, 256), b256, 0, stream>>>(xdbl, xp, xz, dt_w + dir*4*64,
              dt_b + dir*64, Alog + dir*64*64, Dp + dir*64, y16, fwd);
      // Yacc += y @ out_w[dir]                   (16384 x 64 x 128)
      k_gemm<<<dim3(256, 2), b256, 0, stream>>>(y16, 64, ow16 + dir*64*128, 128,
              Yacc, Yacc, nullptr, 128, NT_, 128, 64, 1.0f, 1.0f);
    }
    k_ln<<<cdiv(NT_*32, 256), b256, 0, stream>>>(Yacc, nullptr, nullptr,
            ln1_g + i*128, ln1_b + i*128, Y3, Y316);
    k_gemm<<<dim3(256, 1), b256, 0, stream>>>(Y316, 128, w1_16 + i*128*32, 32,
            nullptr, F1, nullptr, 32, NT_, 32, 128, 1.0f, 0.0f);
    k_bias_relu<<<cdiv(NT_*32, 256), b256, 0, stream>>>(F1, ffn_b1 + i*32, G16, NT_*32);
    k_gemm<<<dim3(256, 2), b256, 0, stream>>>(G16, 32, w2_16 + i*32*128, 128,
            nullptr, F2, nullptr, 128, NT_, 128, 32, 1.0f, 0.0f);
    // x = LN2(F2 + b2 + Y3)  (also writes f16 mirror for next round)
    k_ln<<<cdiv(NT_*32, 256), b256, 0, stream>>>(F2, Y3, ffn_b2 + i*128,
            ln2_g + i*128, ln2_b + i*128, x_cur, x16);
  }

  // ---------------- head ----------------
  k_transpose<<<cdiv(128*NT_, 256), b256, 0, stream>>>(x_cur, Hf, H16);
  k_Ag  <<<128, 128, 0, stream>>>(psi_emb, psi_s, Ag);
  k_attn<<<512, 128, 0, stream>>>(psi_emb, W_q, W_k, alpha_s, Ag, Aeff, Aeff16);
  k_Wf  <<<cdiv(4*128*4*512, 256), b256, 0, stream>>>(psi_emb, F_w, Wf);
  k_zero<<<cdiv(32*128, 256), b256, 0, stream>>>(outb, 32*128);

  for (int h = 0; h < 4; ++h) {
    const h16* Ah = Aeff16 + h * 128 * 128;
    // T1 = A @ H
    k_gemm<<<dim3(2, 256), b256, 0, stream>>>(Ah, 128, H16, NT_,
            nullptr, T1, T1h, NT_, 128, NT_, 128, 1.0f, 0.0f);
    // T2 = 2*A @ T1 - H
    k_gemm<<<dim3(2, 256), b256, 0, stream>>>(Ah, 128, T1h, NT_,
            Hf, T2, T2h, NT_, 128, NT_, 128, 2.0f, -1.0f);
    // T3 = 2*A @ T2 - T1
    k_gemm<<<dim3(2, 256), b256, 0, stream>>>(Ah, 128, T2h, NT_,
            T1, T3, nullptr, NT_, 128, NT_, 128, 2.0f, -1.0f);
    k_head_reduce<<<cdiv(128*32*32, 256), b256, 0, stream>>>(Hf, T1, T2, T3, Wf,
            psi_emb, f_b, hmix, outb, h);
  }

  // pred = out @ head_w + head_b ; log_var = zeros(32,128)
  k_pred<<<4, 256, 0, stream>>>(outb, head_w, head_b, outp);
  k_zero<<<cdiv(4096, 256), b256, 0, stream>>>(outp + 32, 4096);
}
